// EdgeBlock_30176440222299
// MI455X (gfx1250) — matrix operand order repaired and checked by reading, not compiled, not run on hardware
//
#include <hip/hip_runtime.h>
#include <hip/hip_bf16.h>

// ---------------------------------------------------------------------------
// EdgeConv (DGCNN) block for MI455X / gfx1250, wave32 + WMMA + async-to-LDS.
//   Stage A: per-point squared norms
//   Stage B: bf16 weight conversion (weights stay resident in 192MB L2)
//   Stage C: kNN: candidate tiles staged via GLOBAL_LOAD_ASYNC_TO_LDS_B128,
//            distances via V_WMMA_F32_16X16X4_F32, per-row sorted top-21
//   Stage D: fully fused gather -> 4 GEMMs (V_WMMA_F32_16X16X32_BF16, A-frags
//            register-resident per row-tile) -> softmax -> weighted reduction
// ---------------------------------------------------------------------------

#define BATCH 4
#define NPTS  4096
#define FIN   128
#define FOUT  256
#define KNN   20
#define SLOPE 0.2f

#define PT    4            // points per fused workgroup
#define COLS  (PT * KNN)   // 80 = 5 WMMA column tiles

typedef __attribute__((ext_vector_type(16))) __bf16 v16bf;
typedef __attribute__((ext_vector_type(8)))  __bf16 v8bf;
typedef __attribute__((ext_vector_type(8)))  float  v8f;
typedef __attribute__((ext_vector_type(2)))  float  v2f;

// ---------------------------------------------------------------------------
// Stage A: squared norms per point
// ---------------------------------------------------------------------------
__global__ __launch_bounds__(128) void sqnorm_kernel(const float* __restrict__ x,
                                                     float* __restrict__ sq) {
  const int p = blockIdx.x;          // b*N + n
  const int t = threadIdx.x;         // 128 = FIN
  __shared__ float red[128];
  float v = x[(size_t)p * FIN + t];
  red[t] = v * v;
  __syncthreads();
  for (int s = 64; s > 0; s >>= 1) {
    if (t < s) red[t] += red[t + s];
    __syncthreads();
  }
  if (t == 0) sq[p] = red[0];
}

// ---------------------------------------------------------------------------
// Stage B: fp32 -> bf16 weight conversion
// ---------------------------------------------------------------------------
__global__ __launch_bounds__(256) void cvt_bf16_kernel(const float* __restrict__ src,
                                                       __bf16* __restrict__ dst, int n) {
  int i = blockIdx.x * 256 + threadIdx.x;
  if (i < n) dst[i] = (__bf16)src[i];
}

// ---------------------------------------------------------------------------
// Stage C: kNN.  256 threads = 8 waves; each wave owns 16 query rows.
// Candidate tile (16 pts x 128 feat fp32, 8KB) is staged into LDS once per
// step with async global->LDS copies and shared by all 8 waves.
// dist = |a|^2 + |b|^2 - 2 a.b ;  top-21 kept sorted per row (lane 0..15).
// ---------------------------------------------------------------------------
__global__ __launch_bounds__(256) void topk_kernel(const float* __restrict__ x,
                                                   const float* __restrict__ sq,
                                                   int* __restrict__ idxout) {
  const int tid  = threadIdx.x;
  const int wave = tid >> 5;
  const int lane = tid & 31;
  const int l16  = lane & 15;
  const int half = lane >> 4;

  const int blocksPerBatch = NPTS / 128;               // 32
  const int b     = blockIdx.x / blocksPerBatch;
  const int rbase = (blockIdx.x % blocksPerBatch) * 128 + wave * 16;

  const float* xb  = x  + (size_t)b * NPTS * FIN;
  const float* sqb = sq + (size_t)b * NPTS;

  __shared__ float cand[16][132];                      // 528B row stride (16B skew)
  __shared__ float tile[8][16][17];                    // per-wave dist tile

  // per-row sorted top-21 (distance ascending); slot 0 ends up == self
  float bd[21];
  int   bi[21];
  for (int i = 0; i < 21; ++i) { bd[i] = 3.4e38f; bi[i] = 0; }
  float dmax = 3.4e38f;                                // == bd[20], register copy

  // row |.|^2 per accumulator VGPR:  M = v + 8*half
  float sm[8];
#pragma unroll
  for (int v = 0; v < 8; ++v) sm[v] = sqb[rbase + v + 8 * half];

  // A operand (query rows), loop-invariant: 32 x v2f held in VGPRs.
  // f32 A 16x4 layout: lane<16 -> K{0,1}; lane>=16 -> K{2,3}.
  const float* ap = xb + (size_t)(rbase + l16) * FIN + 2 * half;
  v2f A[32];
#pragma unroll
  for (int k = 0; k < 32; ++k) A[k] = *(const v2f*)(ap + 4 * k);

  for (int c0 = 0; c0 < NPTS; c0 += 16) {
    __syncthreads();   // previous candidate tile fully consumed

    // ---- async global->LDS staging: 512 x b128 chunks, 2 per thread
#pragma unroll
    for (int it = 0; it < 2; ++it) {
      const int  chunk = tid + it * 256;               // 0..511
      const int  row   = chunk >> 5;                   // candidate row 0..15
      const int  part  = chunk & 31;                   // 16B chunk in row
      unsigned long long ga =
          (unsigned long long)(xb + (size_t)(c0 + row) * FIN + part * 4);
      unsigned lo = (unsigned)(size_t)(&cand[row][part * 4]);
      asm volatile("global_load_async_to_lds_b128 %0, %1, off"
                   :: "v"(lo), "v"(ga) : "memory");
    }
    asm volatile("s_wait_asynccnt 0" ::: "memory");
    __syncthreads();                                    // tile visible to all waves

    // ---- B fragments from LDS (short latency), then back-to-back WMMAs
    const float* cp = &cand[l16][2 * half];
    v2f Bv[32];
#pragma unroll
    for (int k = 0; k < 32; ++k) Bv[k] = *(const v2f*)(cp + 4 * k);

    v8f acc = {};
#pragma unroll
    for (int k = 0; k < 32; ++k)
      acc = __builtin_amdgcn_wmma_f32_16x16x4_f32(false, A[k], false, Bv[k],
                                                  (short)0, acc, false, false);

    const float sqc = sqb[c0 + l16];                   // column |.|^2 (N = l16)
#pragma unroll
    for (int v = 0; v < 8; ++v)
      tile[wave][v + 8 * half][l16] = sm[v] + sqc - 2.0f * acc[v];

    // wave-local LDS visibility (tile is per-wave; no s_barrier needed)
    asm volatile("s_wait_dscnt 0" ::: "memory");

    if (half == 0) {
      for (int j = 0; j < 16; ++j) {
        float d = tile[wave][l16][j];
        if (d < dmax) {
          int ci = c0 + j;
          int p = 20;
          while (p > 0 && bd[p - 1] > d) {
            bd[p] = bd[p - 1]; bi[p] = bi[p - 1]; --p;
          }
          bd[p] = d; bi[p] = ci;
          dmax = bd[20];
        }
      }
    }
  }

  if (half == 0) {
    int* op = idxout + ((size_t)b * NPTS + rbase + l16) * KNN;
    for (int k = 0; k < KNN; ++k) op[k] = bi[k + 1];   // drop self
  }
}

// ---------------------------------------------------------------------------
// bf16 GEMM helpers.  A fragments (weights, row-major global) are loaded once
// per row-tile and kept in VGPRs; B fragments stream from column-major LDS.
//   bf16 A 16x32 layout/lane: K in {ka..ka+7, ka+16..ka+23}, ka = 8*half
//   bf16 B 32x16 layout/lane: K in {kb..kb+7, kb+16..kb+23} of column l16, kb = 8*half
//   f32  C/D layout: element (M = v + 8*half, N = l16) in acc[v]
// ---------------------------------------------------------------------------
template <int NK>
__device__ __forceinline__ void load_a_tiles(const __bf16* __restrict__ W, int ldw,
                                             int row0, v16bf* a) {
  const int lane = threadIdx.x & 31;
  const int l16 = lane & 15, half = lane >> 4;
  const __bf16* wrow = W + (size_t)(row0 + l16) * ldw + (half ? 8 : 0);
#pragma unroll
  for (int c = 0; c < NK; ++c) {
    v8bf a0 = *(const v8bf*)(wrow + c * 32);
    v8bf a1 = *(const v8bf*)(wrow + c * 32 + 16);
    v16bf t;
#pragma unroll
    for (int i = 0; i < 8; ++i) { t[i] = a0[i]; t[i + 8] = a1[i]; }
    a[c] = t;
  }
}

template <int NK>
__device__ __forceinline__ v8f gemm_cols(const v16bf* a, const __bf16* E, int lde,
                                         int col0, int koff, v8f acc) {
  const int lane = threadIdx.x & 31;
  const int l16 = lane & 15, half = lane >> 4;
  const __bf16* ecol = E + (col0 + l16) * lde + koff + (half ? 8 : 0);
#pragma unroll
  for (int c = 0; c < NK; ++c) {
    v8bf b0 = *(const v8bf*)(ecol + c * 32);
    v8bf b1 = *(const v8bf*)(ecol + c * 32 + 16);
    v16bf bm;
#pragma unroll
    for (int i = 0; i < 8; ++i) { bm[i] = b0[i]; bm[i + 8] = b1[i]; }
    acc = __builtin_amdgcn_wmma_f32_16x16x32_bf16(false, a[c], false, bm,
                                                  (short)0, acc, false, false);
  }
  return acc;
}

__device__ __forceinline__ v8f bias_init(const float* __restrict__ bias, int row0) {
  const int half = (threadIdx.x & 31) >> 4;
  v8f acc;
#pragma unroll
  for (int v = 0; v < 8; ++v) acc[v] = bias[row0 + v + 8 * half];
  return acc;
}

// ---------------------------------------------------------------------------
// Stage D: fused gather + 4 GEMMs + softmax + weighted sum for PT points.
// LDS layout (dynamic, strides padded to kill 256B bank periodicity):
//   E   : 256 x 80 bf16  col-major, stride 264  ->  42240 B
//   T1w : 128 x 80 bf16  col-major, stride 136  ->  21760 B
//   T1x : 256 x 80 bf16  col-major, stride 264  ->  42240 B
//   S   : 256 x 80 f32   col-major, stride 260  ->  83200 B
//   oac : 256 x PT f32                          ->   4096 B
// total 193536 B  ( < 320 KB WGP LDS )
// ---------------------------------------------------------------------------
#define E_STRIDE   264
#define T1W_STRIDE 136
#define T1X_STRIDE 264
#define S_STRIDE   260
#define LDS_BYTES  193536

__global__ __launch_bounds__(256) void fused_conv_kernel(
    const float* __restrict__ x, const int* __restrict__ nidx,
    const __bf16* __restrict__ w1, const float* __restrict__ b1,
    const __bf16* __restrict__ w2, const float* __restrict__ b2,
    const __bf16* __restrict__ x1w, const float* __restrict__ b3,
    const __bf16* __restrict__ x2w, const float* __restrict__ b4,
    float* __restrict__ out) {
  extern __shared__ char smem[];
  __bf16* E   = (__bf16*)(smem);
  __bf16* T1w = (__bf16*)(smem + 42240);
  __bf16* T1x = (__bf16*)(smem + 64000);
  float*  S   = (float*)(smem + 106240);
  float*  oac = (float*)(smem + 189440);
  __shared__ int nb[COLS];

  const int tid  = threadIdx.x;
  const int wave = tid >> 5;
  const int lane = tid & 31;
  const int l16  = lane & 15;
  const int half = lane >> 4;

  const int b  = blockIdx.x / (NPTS / PT);
  const int n0 = (blockIdx.x % (NPTS / PT)) * PT;
  const float* xb = x + (size_t)b * NPTS * FIN;

  if (tid < COLS)
    nb[tid] = nidx[((size_t)b * NPTS + n0 + tid / KNN) * KNN + (tid % KNN)];
  for (int i = tid; i < FOUT * PT; i += 256) oac[i] = 0.0f;
  __syncthreads();

  // ---- gather edge features:  E[0:128] = central, E[128:256] = nbr-central
  for (int i = tid; i < COLS * FIN; i += 256) {
    const int c = i >> 7;                // column (point p, neighbor k)
    const int r = i & 127;               // feature row
    const int p = c / KNN;
    float cen = xb[(size_t)(n0 + p) * FIN + r];
    float ngh = xb[(size_t)nb[c] * FIN + r];
    E[c * E_STRIDE + r]       = (__bf16)cen;
    E[c * E_STRIDE + 128 + r] = (__bf16)(ngh - cen);
  }
  __syncthreads();

  // ---- w-branch GEMM1: T1w = lrelu(w1 * E[128:256] + b1)   (128 x 80)
  {
    const int row0 = wave * 16;          // 8 row tiles / 8 waves
    v16bf a[4];
    load_a_tiles<4>(w1, FIN, row0, a);
    const v8f binit = bias_init(b1, row0);
    for (int ct = 0; ct < 5; ++ct) {
      v8f acc = gemm_cols<4>(a, E, E_STRIDE, ct * 16, 128, binit);
      v8bf pk;
#pragma unroll
      for (int v = 0; v < 8; ++v) {
        float f = acc[v];
        pk[v] = (__bf16)(f > 0.0f ? f : SLOPE * f);
      }
      *(v8bf*)(T1w + (ct * 16 + l16) * T1W_STRIDE + row0 + 8 * half) = pk;
    }
  }
  __syncthreads();

  // ---- w-branch GEMM2: S = w2 * T1w + b2   (256 x 80, fp32)
  for (int rt = wave; rt < 16; rt += 8) {
    const int row0 = rt * 16;
    v16bf a[4];
    load_a_tiles<4>(w2, 128, row0, a);
    const v8f binit = bias_init(b2, row0);
    for (int ct = 0; ct < 5; ++ct) {
      v8f acc = gemm_cols<4>(a, T1w, T1W_STRIDE, ct * 16, 0, binit);
      float* sp = S + (ct * 16 + l16) * S_STRIDE + row0 + 8 * half;
      *(float4*)(sp)     = make_float4(acc[0], acc[1], acc[2], acc[3]);
      *(float4*)(sp + 4) = make_float4(acc[4], acc[5], acc[6], acc[7]);
    }
  }
  __syncthreads();

  // ---- softmax over K=20 per (o, p), in place in S (fp32)
  for (int i = tid; i < FOUT * PT; i += 256) {
    const int o = i & 255, p = i >> 8;
    float vals[KNN], mx = -3.4e38f;
#pragma unroll
    for (int k = 0; k < KNN; ++k) {
      float v = S[(p * KNN + k) * S_STRIDE + o];
      vals[k] = v; mx = fmaxf(mx, v);
    }
    float sum = 0.0f;
#pragma unroll
    for (int k = 0; k < KNN; ++k) { float e = __expf(vals[k] - mx); vals[k] = e; sum += e; }
    const float inv = 1.0f / sum;
#pragma unroll
    for (int k = 0; k < KNN; ++k) S[(p * KNN + k) * S_STRIDE + o] = vals[k] * inv;
  }
  __syncthreads();

  // ---- x-branch GEMM1: T1x = lrelu(x1 * E + b3)   (256 x 80)
  for (int rt = wave; rt < 16; rt += 8) {
    const int row0 = rt * 16;
    v16bf a[8];
    load_a_tiles<8>(x1w, 2 * FIN, row0, a);
    const v8f binit = bias_init(b3, row0);
    for (int ct = 0; ct < 5; ++ct) {
      v8f acc = gemm_cols<8>(a, E, E_STRIDE, ct * 16, 0, binit);
      v8bf pk;
#pragma unroll
      for (int v = 0; v < 8; ++v) {
        float f = acc[v];
        pk[v] = (__bf16)(f > 0.0f ? f : SLOPE * f);
      }
      *(v8bf*)(T1x + (ct * 16 + l16) * T1X_STRIDE + row0 + 8 * half) = pk;
    }
  }
  __syncthreads();

  // ---- x-branch GEMM2 fused with weighted reduction over K
  for (int rt = wave; rt < 16; rt += 8) {
    const int row0 = rt * 16;
    v16bf a[8];
    load_a_tiles<8>(x2w, FOUT, row0, a);
    const v8f binit = bias_init(b4, row0);
    for (int ct = 0; ct < 5; ++ct) {
      v8f acc = gemm_cols<8>(a, T1x, T1X_STRIDE, ct * 16, 0, binit);
      const int c = ct * 16 + l16;
      const int p = c / KNN;
#pragma unroll
      for (int v = 0; v < 8; ++v) {
        const int o = row0 + v + 8 * half;
        atomicAdd(&oac[o * PT + p], acc[v] * S[c * S_STRIDE + o]);
      }
    }
  }
  __syncthreads();

  // ---- write out[b, n0+p, o]  (B, N, 256) fp32
  for (int i = tid; i < FOUT * PT; i += 256) {
    const int o = i & 255, p = i >> 8;
    out[((size_t)b * NPTS + n0 + p) * FOUT + o] = oac[o * PT + p];
  }
}

// ---------------------------------------------------------------------------
// Host launcher
// ---------------------------------------------------------------------------
extern "C" void kernel_launch(void* const* d_in, const int* in_sizes, int n_in,
                              void* d_out, int out_size, void* d_ws, size_t ws_size,
                              hipStream_t stream) {
  const float* x   = (const float*)d_in[0];
  const float* w1w = (const float*)d_in[1];
  const float* w1b = (const float*)d_in[2];
  const float* w2w = (const float*)d_in[3];
  const float* w2b = (const float*)d_in[4];
  const float* x1w = (const float*)d_in[5];
  const float* x1b = (const float*)d_in[6];
  const float* x2w = (const float*)d_in[7];
  const float* x2b = (const float*)d_in[8];
  float* out = (float*)d_out;

  char* ws = (char*)d_ws;
  float*  sq    = (float*)(ws);                               //   64 KB
  int*    idx   = (int*)(ws + 65536);                         // 1.25 MB
  __bf16* w1bf  = (__bf16*)(ws + 1376256);                    //   32 KB
  __bf16* w2bf  = (__bf16*)(ws + 1409024);                    //   64 KB
  __bf16* x1bf  = (__bf16*)(ws + 1474560);                    //  128 KB
  __bf16* x2bf  = (__bf16*)(ws + 1605632);                    //  128 KB

  // Stage A: squared norms
  sqnorm_kernel<<<BATCH * NPTS, 128, 0, stream>>>(x, sq);

  // Stage B: bf16 weights (stay hot in 192MB L2 for all fused WGs)
  cvt_bf16_kernel<<<(128 * 128 + 255) / 256, 256, 0, stream>>>(w1w, w1bf, 128 * 128);
  cvt_bf16_kernel<<<(256 * 128 + 255) / 256, 256, 0, stream>>>(w2w, w2bf, 256 * 128);
  cvt_bf16_kernel<<<(256 * 256 + 255) / 256, 256, 0, stream>>>(x1w, x1bf, 256 * 256);
  cvt_bf16_kernel<<<(256 * 256 + 255) / 256, 256, 0, stream>>>(x2w, x2bf, 256 * 256);

  // Stage C: kNN (128 rows per 256-thread block, async-staged candidate tiles)
  topk_kernel<<<BATCH * (NPTS / 128), 256, 0, stream>>>(x, sq, idx);

  // Stage D: fused edge-conv, one WG per PT points, 189KB dynamic LDS
  fused_conv_kernel<<<BATCH * (NPTS / PT), 256, LDS_BYTES, stream>>>(
      x, idx, w1bf, w1b, w2bf, w2b, x1bf, x1b, x2bf, x2b, out);
}
